// LIMITR_13460427506290
// MI455X (gfx1250) — matrix operand order repaired and checked by reading, not compiled, not run on hardware
//
#include <hip/hip_runtime.h>
#include <hip/hip_bf16.h>

// ---------------------------------------------------------------------------
// Problem constants (from reference): B=Bw=32, C=1024, D=1000 (padded 1024),
// H*W=256, S=512, Q=64.
// ---------------------------------------------------------------------------
#define BATCH     32
#define CCH       1024
#define DP        1024      // padded D
#define DREAL     1000
#define SPA       256       // H*W
#define SS        512       // concat spatial
#define QQ        64

typedef __attribute__((ext_vector_type(16))) __bf16 bf16x16;
typedef __attribute__((ext_vector_type(8)))  float  floatx8;

union FragBF { unsigned u[8]; bf16x16 v; };

__device__ __forceinline__ __bf16 f2bf(float f) {
  unsigned u = __builtin_bit_cast(unsigned, f);
  unsigned r = (u + 0x7FFFu + ((u >> 16) & 1u)) >> 16;
  unsigned short s = (unsigned short)r;
  return __builtin_bit_cast(__bf16, s);
}

// Raw CDNA5 workgroup split barrier.  Unlike __syncthreads(), this does NOT
// force a full s_wait_loadcnt 0x0 drain, so next-chunk A-fragment global
// loads stay in flight across the LDS tile hand-off.
__device__ __forceinline__ void wg_barrier() {
  asm volatile("s_barrier_signal -1\n\ts_barrier_wait -1" ::: "memory");
}

// ---------------------------------------------------------------------------
// Generic batched bf16 NT GEMM:  C[m,n] = sum_k A[m,k] * Bt[n,k]
// A row-major [M,K] bf16, Bt row-major [N,K] bf16 (K contiguous), C f32.
// Block = (32,4): 4 waves, one 16x64 C tile each (4 consecutive M tiles,
// shared N tile).  B tile (64 rows x 32 K) is staged once per block into a
// double-buffered LDS tile via CDNA5 async copies
// (global_load_async_to_lds_b128, ASYNCcnt), while A fragments are
// double-buffered in registers, so the 4 v_wmma_f32_16x16x32_bf16 per step
// overlap the next step's data movement.
// Batch indexing: Aoff = ((z/aDiv)%aMod)*sA, Boff = ((z/bDiv)%bMod)*sB.
// EPI: 0=none, 1=C+=bias[m], 2=C=tanh(C+bias[n]).
// NOTE: all launches use M exactly divisible (no partial blocks), so the
// early-out never fires and barrier participation is uniform.
// ---------------------------------------------------------------------------
template<int EPI>
__global__ __launch_bounds__(128)
void gemm_bf16_nt(const __bf16* __restrict__ A, long sA, int aDiv, int aMod, int lda,
                  const __bf16* __restrict__ Bt, long sB, int bDiv, int bMod, int ldb,
                  float* __restrict__ C, long sC, int ldc,
                  int M, int K, const float* __restrict__ bias, int biasLen)
{
  __shared__ __align__(16) __bf16 btile[2 * 64 * 32];   // 2 x 4 KB double buffer

  const int lane = threadIdx.x;                       // 0..31 (wave32)
  const int tM = (blockIdx.y * blockDim.y + threadIdx.y) << 4;
  if (tM >= M) return;                                // never taken (see note)
  const int tN = blockIdx.x << 6;
  const int z  = blockIdx.z;

  A  += (long)((z / aDiv) % aMod) * sA + (long)tM * lda;
  Bt += (long)((z / bDiv) % bMod) * sB + (long)tN * ldb;
  C  += (long)z * sC;

  const int m16 = lane & 15;
  const int hi  = lane >> 4;
  const int kbA = hi * 16;                // A: lanes 16-31 take K+16..31

  const __bf16* aRow = A + (long)m16 * lda;
  const unsigned ldsBase = (unsigned)(size_t)(void*)btile;   // LDS byte offset
  const int tFlat = threadIdx.y * 32 + lane;                 // 0..127

  // Issue the async copy of one 64x32 bf16 B tile (4 KB) into LDS buffer buf.
  // 256 16-byte chunks, 2 per thread -> ASYNCcnt += 2 per wave issue.
  auto issueB = [&](int buf, int kk) {
    #pragma unroll
    for (int c = 0; c < 2; ++c) {
      const int chunk = tFlat + c * 128;            // 0..255
      const int row   = chunk >> 2;                 // B tile row (n)
      const int part  = chunk & 3;                  // 16-byte part within row
      const __bf16* g = Bt + (long)row * ldb + kk + part * 8;
      const unsigned lo = ldsBase + (unsigned)(buf * 4096 + row * 64 + part * 16);
      const unsigned long long ga = (unsigned long long)(size_t)g;
      asm volatile("global_load_async_to_lds_b128 %0, %1, off"
                   :: "v"(lo), "v"(ga) : "memory");
    }
  };

  // Load a 16x32 A fragment (contiguous k: lane half h holds K 16h..16h+15) from global into registers.
  auto loadA = [&](FragBF& fa, int kk) {
    const unsigned* pa = (const unsigned*)(aRow + kk);
    #pragma unroll
    for (int v = 0; v < 4; ++v) fa.u[v]     = pa[(kbA >> 1) + v];
    #pragma unroll
    for (int v = 0; v < 4; ++v) fa.u[4 + v] = pa[4 + (kbA >> 1) + v];
    __builtin_prefetch((const void*)(aRow + kk + 64), 0, 1);   // global_prefetch_b8
  };

  // Read one 32x16 B fragment from the LDS tile (lanes 0-15: K 0-15,
  // lanes 16-31: K 16-31), rows nrow..: 32 contiguous bytes -> ds_load_b128 x2.
  auto readB = [&](FragBF& fb, int buf, int nrow) {
    const unsigned* p = (const unsigned*)(btile + buf * 2048 + nrow * 32 + hi * 16);
    #pragma unroll
    for (int v = 0; v < 8; ++v) fb.u[v] = p[v];
  };

  floatx8 c0 = {}, c1 = {}, c2 = {}, c3 = {};
  FragBF faA, faB;

  // ---- pipeline prologue: stage K chunk 0
  issueB(0, 0);
  loadA(faA, 0);

  for (int kk = 0; kk < K; kk += 64) {
    // ======== phase 0: compute on buf0/faA (chunk kk), stage chunk kk+32 ====
    issueB(1, kk + 32);
    loadA(faB, kk + 32);
    asm volatile("s_wait_asynccnt 0x2" ::: "memory");   // buf0 copies done
    wg_barrier();
    {
      FragBF fb0, fb1, fb2, fb3;
      readB(fb0, 0, m16); readB(fb1, 0, m16 + 16);
      readB(fb2, 0, m16 + 32); readB(fb3, 0, m16 + 48);
      c0 = __builtin_amdgcn_wmma_f32_16x16x32_bf16(false, faA.v, false, fb0.v, (short)0, c0, false, false);
      c1 = __builtin_amdgcn_wmma_f32_16x16x32_bf16(false, faA.v, false, fb1.v, (short)0, c1, false, false);
      c2 = __builtin_amdgcn_wmma_f32_16x16x32_bf16(false, faA.v, false, fb2.v, (short)0, c2, false, false);
      c3 = __builtin_amdgcn_wmma_f32_16x16x32_bf16(false, faA.v, false, fb3.v, (short)0, c3, false, false);
    }
    wg_barrier();                                       // buf0 free for reuse

    // ======== phase 1: compute on buf1/faB (chunk kk+32), stage chunk kk+64 =
    if (kk + 64 < K) {
      issueB(0, kk + 64);
      loadA(faA, kk + 64);
      asm volatile("s_wait_asynccnt 0x2" ::: "memory"); // buf1 copies done
    } else {
      asm volatile("s_wait_asynccnt 0x0" ::: "memory");
    }
    wg_barrier();
    {
      FragBF fb0, fb1, fb2, fb3;
      readB(fb0, 1, m16); readB(fb1, 1, m16 + 16);
      readB(fb2, 1, m16 + 32); readB(fb3, 1, m16 + 48);
      c0 = __builtin_amdgcn_wmma_f32_16x16x32_bf16(false, faB.v, false, fb0.v, (short)0, c0, false, false);
      c1 = __builtin_amdgcn_wmma_f32_16x16x32_bf16(false, faB.v, false, fb1.v, (short)0, c1, false, false);
      c2 = __builtin_amdgcn_wmma_f32_16x16x32_bf16(false, faB.v, false, fb2.v, (short)0, c2, false, false);
      c3 = __builtin_amdgcn_wmma_f32_16x16x32_bf16(false, faB.v, false, fb3.v, (short)0, c3, false, false);
    }
    wg_barrier();
  }

  // ---- Epilogue + store.  lane -> col tN + (lane&15) (+16/32/48),
  //      VGPR v -> row tM + hi*8 + v.
  const int nc = tN + m16;
  #pragma unroll
  for (int v = 0; v < 8; ++v) {
    const int m = tM + hi * 8 + v;
    float x0 = c0[v], x1 = c1[v], x2 = c2[v], x3 = c3[v];
    if (EPI == 1) {
      const float bm = (m < biasLen) ? bias[m] : 0.f;
      x0 += bm; x1 += bm; x2 += bm; x3 += bm;
    }
    if (EPI == 2) {
      x0 = tanhf(x0 + ((nc      < biasLen) ? bias[nc]      : 0.f));
      x1 = tanhf(x1 + ((nc + 16 < biasLen) ? bias[nc + 16] : 0.f));
      x2 = tanhf(x2 + ((nc + 32 < biasLen) ? bias[nc + 32] : 0.f));
      x3 = tanhf(x3 + ((nc + 48 < biasLen) ? bias[nc + 48] : 0.f));
    }
    float* cr = C + (long)m * ldc + nc;
    cr[0] = x0; cr[16] = x1; cr[32] = x2; cr[48] = x3;
  }
}

// ---------------------------------------------------------------------------
// Prep / conversion kernels
// ---------------------------------------------------------------------------
__global__ void k_cvt_convw(const float* __restrict__ w, __bf16* __restrict__ o) {
  long t = (long)blockIdx.x * 256 + threadIdx.x;      // DP*CCH
  int m = (int)(t >> 10), c = (int)(t & 1023);
  o[t] = (m < DREAL) ? f2bf(w[(long)m * CCH + c]) : f2bf(0.f);
}

// featT[b][s][c]: s<256 -> frontal, else lateral
__global__ void k_featT(const float* __restrict__ f, const float* __restrict__ l,
                        __bf16* __restrict__ o) {
  long t = (long)blockIdx.x * 256 + threadIdx.x;      // BATCH*SS*CCH
  int b = (int)(t >> 19);
  int rem = (int)(t & 524287);
  int s = rem >> 10, c = rem & 1023;
  float v = (s < SPA) ? f[(long)b * CCH * SPA + (long)c * SPA + s]
                      : l[(long)b * CCH * SPA + (long)c * SPA + (s - SPA)];
  o[t] = f2bf(v);
}

// wordsT[i][q][d] (zero-padded d)
__global__ void k_wordsT(const float* __restrict__ w, __bf16* __restrict__ o) {
  long t = (long)blockIdx.x * 256 + threadIdx.x;      // BATCH*QQ*DP
  int i = (int)(t >> 16);
  int rem = (int)(t & 65535);
  int q = rem >> 10, d = rem & 1023;
  o[t] = (d < DREAL) ? f2bf(w[(long)i * DREAL * QQ + (long)d * QQ + q]) : f2bf(0.f);
}

// saT[e][d] = sa[d][e] (zero padded)
__global__ void k_saT(const float* __restrict__ src, __bf16* __restrict__ o) {
  long t = (long)blockIdx.x * 256 + threadIdx.x;      // DP*DP
  int e = (int)(t >> 10), d = (int)(t & 1023);
  o[t] = (d < DREAL && e < DREAL) ? f2bf(src[(long)d * DREAL + e]) : f2bf(0.f);
}

// img f32 [b,d,s] -> img_bf [b,d,s] and imgT_bf [b,s,d]
__global__ void k_imgcvt(const float* __restrict__ img, __bf16* __restrict__ obf,
                         __bf16* __restrict__ otr) {
  long t = (long)blockIdx.x * 256 + threadIdx.x;      // BATCH*DP*SS
  int b = (int)(t >> 19);
  int rem = (int)(t & 524287);
  int d = rem >> 9, s = rem & 511;
  __bf16 v = f2bf(img[t]);
  obf[t] = v;
  otr[(long)b * 524288 + (long)s * 1024 + d] = v;
}

// ---------------------------------------------------------------------------
// In-place wave softmax over rows of 64 (wave32: 2 elems/lane)
// ---------------------------------------------------------------------------
__global__ void k_softmax64(float* __restrict__ p, int rows) {
  int wave = (blockIdx.x * blockDim.x + threadIdx.x) >> 5;
  int lane = threadIdx.x & 31;
  if (wave >= rows) return;
  float* r = p + (long)wave * 64;
  float a = r[lane], b = r[lane + 32];
  float m = fmaxf(a, b);
  #pragma unroll
  for (int off = 16; off; off >>= 1) m = fmaxf(m, __shfl_xor(m, off, 32));
  float ea = __expf(a - m), eb = __expf(b - m);
  float s = ea + eb;
  #pragma unroll
  for (int off = 16; off; off >>= 1) s += __shfl_xor(s, off, 32);
  float inv = 1.0f / s;
  r[lane] = ea * inv; r[lane + 32] = eb * inv;
}

__device__ __forceinline__ float blockSum(float v, float* sh) {
  int tid = threadIdx.x;
  sh[tid] = v; __syncthreads();
  #pragma unroll
  for (int s = 128; s > 0; s >>= 1) { if (tid < s) sh[tid] += sh[tid + s]; __syncthreads(); }
  float r = sh[0]; __syncthreads();
  return r;
}

// a2[i,b,q,s] = softmax_s( 4 * a1[i,b,s,q] ); also bf16 copy for wctx GEMM A
__global__ void k_a2(const float* __restrict__ a1, float* __restrict__ a2,
                     __bf16* __restrict__ a2bf) {
  __shared__ float sh[256];
  int r = blockIdx.x;                                 // (i*32+b)*64 + q
  int ib = r >> 6, q = r & 63;
  int tid = threadIdx.x;
  const float* base = a1 + (long)ib * (SS * QQ) + q;
  float v0 = base[(long)tid * QQ] * 4.0f;
  float v1 = base[(long)(tid + 256) * QQ] * 4.0f;
  float m = fmaxf(v0, v1);
  sh[tid] = m; __syncthreads();
  #pragma unroll
  for (int s = 128; s > 0; s >>= 1) { if (tid < s) sh[tid] = fmaxf(sh[tid], sh[tid + s]); __syncthreads(); }
  m = sh[0]; __syncthreads();
  float e0 = __expf(v0 - m), e1 = __expf(v1 - m);
  float tot = blockSum(e0 + e1, sh);
  float inv = 1.0f / tot;
  long o = (long)r * SS;
  a2[o + tid] = e0 * inv; a2[o + tid + 256] = e1 * inv;
  a2bf[o + tid] = f2bf(e0 * inv); a2bf[o + tid + 256] = f2bf(e1 * inv);
}

// att_maps output: out[1024 + i*Q*S + q*S + s] = a2[i,i,q,s]
__global__ void k_att(const float* __restrict__ a2, float* __restrict__ out) {
  long t = (long)blockIdx.x * 256 + threadIdx.x;      // BATCH*QQ*SS
  int i = (int)(t >> 15);
  int rem = (int)(t & 32767);
  out[1024 + t] = a2[(long)(i * 33) * 32768 + rem];   // ib = i*32 + i
}

// sim_loc = l2norm(wctx * word) over d; writes f32 + bf16
__global__ void k_simloc(const float* __restrict__ wctx, const float* __restrict__ words,
                         float* __restrict__ sl, __bf16* __restrict__ slb) {
  __shared__ float sh[256];
  int r = blockIdx.x;                                 // (i*32+b)*64 + q
  int i = r >> 11, q = r & 63;
  int tid = threadIdx.x;
  const float* wrow = wctx + (long)r * DP;
  float t[4]; float ss = 0.f;
  #pragma unroll
  for (int j = 0; j < 4; ++j) {
    int d = tid + j * 256;
    float wd = (d < DREAL) ? words[(long)i * DREAL * QQ + (long)d * QQ + q] : 0.f;
    t[j] = wrow[d] * wd;
    ss += t[j] * t[j];
  }
  float tot = blockSum(ss, sh);
  float inv = 1.0f / (sqrtf(tot) + 1e-8f);
  #pragma unroll
  for (int j = 0; j < 4; ++j) {
    int d = tid + j * 256;
    float v = t[j] * inv;
    sl[(long)r * DP + d] = v;
    slb[(long)r * DP + d] = f2bf(v);
  }
}

// sim_ave[ib][d] = mean_q sim_loc; writes f32 + bf16 (GEMM A for g_emb)
__global__ void k_simave(const float* __restrict__ sl, float* __restrict__ sa,
                         __bf16* __restrict__ sab) {
  int ib = blockIdx.x;
  int tid = threadIdx.x;
  #pragma unroll
  for (int j = 0; j < 4; ++j) {
    int d = tid + j * 256;
    float s = 0.f;
    for (int q = 0; q < QQ; ++q) s += sl[((long)ib * QQ + q) * DP + d];
    float m = s * (1.0f / 64.0f);
    sa[(long)ib * DP + d] = m;
    sab[(long)ib * DP + d] = f2bf(m);
  }
}

// w_raw[ib][q] = sum_d l_emb * g_emb * sa_wc + sa_bc
__global__ void k_w(const float* __restrict__ lemb, const float* __restrict__ gemb,
                    const float* __restrict__ sawc, const float* __restrict__ sabc,
                    float* __restrict__ wraw) {
  __shared__ float sh[256];
  int r = blockIdx.x;                                 // ib*64 + q
  int ib = r >> 6;
  int tid = threadIdx.x;
  float s = 0.f;
  #pragma unroll
  for (int j = 0; j < 4; ++j) {
    int d = tid + j * 256;
    float wc = (d < DREAL) ? sawc[d] : 0.f;
    s += lemb[(long)r * DP + d] * gemb[(long)ib * DP + d] * wc;
  }
  float tot = blockSum(s, sh);
  if (tid == 0) wraw[r] = tot + sabc[0];
}

// new_global = l2norm(sum_q w*sim_loc); sim = sigmoid(ng . cap_w + cap_b); out[b*32+i]
__global__ void k_final(const float* __restrict__ w, const float* __restrict__ sl,
                        const float* __restrict__ capw, const float* __restrict__ capb,
                        float* __restrict__ out) {
  __shared__ float sh[256];
  int ib = blockIdx.x;
  int i = ib >> 5, b = ib & 31;
  int tid = threadIdx.x;
  float ng[4] = {0.f, 0.f, 0.f, 0.f};
  const float* wr = w + (long)ib * QQ;
  const float* sbase = sl + (long)ib * QQ * DP;
  for (int q = 0; q < QQ; ++q) {
    float wq = wr[q];
    const float* row = sbase + (long)q * DP;
    #pragma unroll
    for (int j = 0; j < 4; ++j) ng[j] += wq * row[tid + j * 256];
  }
  float ss = 0.f;
  #pragma unroll
  for (int j = 0; j < 4; ++j) ss += ng[j] * ng[j];
  float tot = blockSum(ss, sh);
  float inv = 1.0f / (sqrtf(tot) + 1e-8f);
  float dot = 0.f;
  #pragma unroll
  for (int j = 0; j < 4; ++j) {
    int d = tid + j * 256;
    dot += ng[j] * inv * ((d < DREAL) ? capw[d] : 0.f);
  }
  float dtot = blockSum(dot, sh);
  if (tid == 0) out[b * 32 + i] = 1.0f / (1.0f + __expf(-(dtot + capb[0])));
}

// ---------------------------------------------------------------------------
// Host launcher
// ---------------------------------------------------------------------------
static inline size_t alignUp256(size_t x) { return (x + 255) & ~(size_t)255; }

extern "C" void kernel_launch(void* const* d_in, const int* in_sizes, int n_in,
                              void* d_out, int out_size, void* d_ws, size_t ws_size,
                              hipStream_t stream) {
  const float* in_l    = (const float*)d_in[0];
  const float* in_f    = (const float*)d_in[1];
  const float* words   = (const float*)d_in[2];
  const float* conv_w  = (const float*)d_in[3];
  const float* conv_b  = (const float*)d_in[4];
  const float* sa_wl   = (const float*)d_in[5];
  const float* sa_bl   = (const float*)d_in[6];
  const float* sa_wg   = (const float*)d_in[7];
  const float* sa_bg   = (const float*)d_in[8];
  const float* sa_wc   = (const float*)d_in[9];
  const float* sa_bc   = (const float*)d_in[10];
  const float* cap_w   = (const float*)d_in[11];
  const float* cap_b   = (const float*)d_in[12];
  float* out = (float*)d_out;

  // ---- workspace layout
  char* ws = (char*)d_ws;
  size_t cur = 0;
  auto take = [&](size_t bytes) -> char* {
    char* p = ws + cur; cur = alignUp256(cur + bytes); return p;
  };
  __bf16* featT_bf  = (__bf16*)take((size_t)BATCH * SS * CCH * 2);   // 33.5 MB
  __bf16* convW_bf  = (__bf16*)take((size_t)DP * CCH * 2);           //  2 MB
  __bf16* wordsT_bf = (__bf16*)take((size_t)BATCH * QQ * DP * 2);    //  4 MB
  __bf16* sawlT_bf  = (__bf16*)take((size_t)DP * DP * 2);            //  2 MB
  __bf16* sawgT_bf  = (__bf16*)take((size_t)DP * DP * 2);            //  2 MB
  float*  img_f32   = (float*) take((size_t)BATCH * DP * SS * 4);    // 67 MB
  __bf16* img_bf    = (__bf16*)take((size_t)BATCH * DP * SS * 2);    // 33.5 MB
  __bf16* imgT_bf   = (__bf16*)take((size_t)BATCH * SS * DP * 2);    // 33.5 MB
  float*  Lbuf      = (float*) take((size_t)BATCH * BATCH * SS * QQ * 4); // 134 MB (logits/a1)
  float*  A2buf     = (float*) take((size_t)BATCH * BATCH * QQ * SS * 4); // 134 MB
  __bf16* A2bf      = (__bf16*)take((size_t)BATCH * BATCH * QQ * SS * 2); //  67 MB
  float*  W1        = (float*) take((size_t)BATCH * BATCH * QQ * DP * 4); // 268 MB (wctx -> l_emb)
  float*  SL        = (float*) take((size_t)BATCH * BATCH * QQ * DP * 4); // 268 MB
  __bf16* SLB       = (__bf16*)take((size_t)BATCH * BATCH * QQ * DP * 2); // 134 MB
  float*  simave    = (float*) take((size_t)BATCH * BATCH * DP * 4);      //   4 MB
  __bf16* simave_bf = (__bf16*)take((size_t)BATCH * BATCH * DP * 2);      //   2 MB
  float*  gemb      = (float*) take((size_t)BATCH * BATCH * DP * 4);      //   4 MB
  float*  wraw      = (float*) take((size_t)BATCH * BATCH * QQ * 4);      // 0.26 MB
  (void)ws_size; (void)in_sizes; (void)n_in; (void)out_size;

  const dim3 gblk(32, 4, 1);   // 4 waves per block

  // ---- Stage 0: conversions / transposes
  k_cvt_convw<<<(DP * CCH) / 256, 256, 0, stream>>>(conv_w, convW_bf);
  k_featT   <<<((size_t)BATCH * SS * CCH) / 256, 256, 0, stream>>>(in_f, in_l, featT_bf);
  k_wordsT  <<<((size_t)BATCH * QQ * DP) / 256, 256, 0, stream>>>(words, wordsT_bf);
  k_saT     <<<(DP * DP) / 256, 256, 0, stream>>>(sa_wl, sawlT_bf);
  k_saT     <<<(DP * DP) / 256, 256, 0, stream>>>(sa_wg, sawgT_bf);

  // ---- G1: 1x1 conv  img[b,o,s] = convW[o,:] . featT[b,s,:] + conv_b[o]
  gemm_bf16_nt<1><<<dim3(SS / 64, (DP / 16) / 4, BATCH), gblk, 0, stream>>>(
      convW_bf, 0L, 1, 1, CCH,
      featT_bf, (long)SS * CCH, 1, BATCH, CCH,
      img_f32, (long)DP * SS, SS, DP, CCH, conv_b, DREAL);

  k_imgcvt<<<((size_t)BATCH * DP * SS) / 256, 256, 0, stream>>>(img_f32, img_bf, imgT_bf);

  // ---- G2: logits[i,b,s,q] = imgT[b,s,:] . wordsT[i,q,:]   (batch z = i*32+b)
  gemm_bf16_nt<0><<<dim3(QQ / 64, (SS / 16) / 4, BATCH * BATCH), gblk, 0, stream>>>(
      imgT_bf, (long)SS * DP, 1, BATCH, DP,
      wordsT_bf, (long)QQ * DP, BATCH, BATCH, DP,
      Lbuf, (long)SS * QQ, QQ, SS, DP, nullptr, 0);

  // ---- a1 = softmax_q(logits) in place; a2 = softmax_s(4*a1^T)
  k_softmax64<<<(BATCH * BATCH * SS) / 8, 256, 0, stream>>>(Lbuf, BATCH * BATCH * SS);
  k_a2<<<BATCH * BATCH * QQ, 256, 0, stream>>>(Lbuf, A2buf, A2bf);
  k_att<<<((size_t)BATCH * QQ * SS) / 256, 256, 0, stream>>>(A2buf, out);

  // ---- G3: wctx[i,b,q,d] = a2[i,b,q,:] . img[b,d,:]   (K = S)
  gemm_bf16_nt<0><<<dim3(DP / 64, 1, BATCH * BATCH), gblk, 0, stream>>>(
      A2bf, (long)QQ * SS, 1, BATCH * BATCH, SS,
      img_bf, (long)DP * SS, 1, BATCH, SS,
      W1, (long)QQ * DP, DP, QQ, SS, nullptr, 0);

  // ---- sim_loc, sim_ave
  k_simloc<<<BATCH * BATCH * QQ, 256, 0, stream>>>(W1, words, SL, SLB);
  k_simave<<<BATCH * BATCH, 256, 0, stream>>>(SL, simave, simave_bf);

  // ---- G4: g_emb = tanh(sim_ave @ sa_wg + sa_bg)   (M = 1024 rows of ib)
  gemm_bf16_nt<2><<<dim3(DP / 64, (BATCH * BATCH / 16) / 4, 1), gblk, 0, stream>>>(
      simave_bf, 0L, 1, 1, DP,
      sawgT_bf, 0L, 1, 1, DP,
      gemb, 0L, DP, BATCH * BATCH, DP, sa_bg, DREAL);

  // ---- G5: l_emb = tanh(sim_loc @ sa_wl + sa_bl)  (M = 65536, reuses W1)
  gemm_bf16_nt<2><<<dim3(DP / 64, (BATCH * BATCH * QQ / 16) / 4, 1), gblk, 0, stream>>>(
      SLB, 0L, 1, 1, DP,
      sawlT_bf, 0L, 1, 1, DP,
      W1, 0L, DP, BATCH * BATCH * QQ, DP, sa_bl, DREAL);

  // ---- w = softmax_q(common . sa_wc + sa_bc); final similarities
  k_w<<<BATCH * BATCH * QQ, 256, 0, stream>>>(W1, gemb, sa_wc, sa_bc, wraw);
  k_softmax64<<<(BATCH * BATCH) / 8, 256, 0, stream>>>(wraw, BATCH * BATCH);
  k_final<<<BATCH * BATCH, 256, 0, stream>>>(wraw, SL, cap_w, cap_b, out);
}
